// ANI2xNoCUAEV_2843268350543
// MI455X (gfx1250) — compile-verified, matrix operand order checked by reading, not run on hardware
//
#include <hip/hip_runtime.h>

typedef __attribute__((ext_vector_type(16))) __bf16 v16bf;
typedef __attribute__((ext_vector_type(8)))  __bf16 v8bf;
typedef __attribute__((ext_vector_type(8)))  float  v8f;

#define NS   7
#define D0P  1024      // AEV (1008) padded to 1024
#define H1S  256
#define H2S  192
#define H3S  160
#define ME   8
#define PI_F 3.14159265358979323846f

// Dynamic LDS layout for mlp_fused (all offsets multiples of 32B)
#define SM_AEV   0        // 16*1024 bf16 = 32768
#define SM_H1    32768    // 8192
#define SM_D1    40960    // 8192
#define SM_H2    49152    // 6144
#define SM_D2    55296    // 6144
#define SM_H3    61440    // 5120
#define SM_D3    66560    // 5120
#define SM_GA    71680    // 8192
#define SM_GB    79872    // 8192
#define SM_ROWS  88064    // 64
#define SM_EP    88128    // 64
#define SM_TOTAL 88192

// ---------------------------------------------------------------- helpers
__device__ __forceinline__ float cutoff(float d, float rc) {
  return d < rc ? 0.5f * __cosf(PI_F * d / rc) + 0.5f : 0.0f;
}
__device__ __forceinline__ float dcutoff(float d, float rc) {
  return d < rc ? -0.5f * (PI_F / rc) * __sinf(PI_F * d / rc) : 0.0f;
}

// 16-row GEMM tile: rowptr = per-lane pointer to row (lane&15) of A (row-major,
// contiguous K), Bt = packed B fragments laid out [kc][lane][16 bf16].
template <int KCN>
__device__ __forceinline__ v8f gemm16(const __bf16* rowptr, const __bf16* Bt,
                                      int lane, v8f c) {
  const int kh = (lane >> 4) * 8;
  for (int kc = 0; kc < KCN; ++kc) {
    const __bf16* pa = rowptr + kc * 32 + kh;
    v8bf lo = *(const v8bf*)(pa);
    v8bf hi = *(const v8bf*)(pa + 16);
    v16bf a;
#pragma unroll
    for (int e = 0; e < 8; ++e) { a[e] = lo[e]; a[e + 8] = hi[e]; }
    v16bf b = *(const v16bf*)(Bt + (size_t)(kc * 32 + lane) * 16);
    c = __builtin_amdgcn_wmma_f32_16x16x32_bf16(false, a, false, b, (short)0, c,
                                                false, false);
  }
  return c;
}

// celu(z,0.1) + derivative, stored bf16 (C frag: row = r + 8*(lane>>4), col = col0+(lane&15))
__device__ __forceinline__ void store_act(v8f c, float bias, __bf16* H,
                                          __bf16* Dv, int ld, int col0, int lane) {
  int rw = (lane >> 4) * 8, cl = col0 + (lane & 15);
#pragma unroll
  for (int r = 0; r < 8; ++r) {
    float z = c[r] + bias;
    float ex = __expf(z * 10.0f);
    float h  = z > 0.0f ? z : 0.1f * (ex - 1.0f);
    float dv = z > 0.0f ? 1.0f : ex;
    H[(rw + r) * ld + cl]  = (__bf16)h;
    Dv[(rw + r) * ld + cl] = (__bf16)dv;
  }
}

__device__ __forceinline__ void store_grad(v8f c, const __bf16* Dv, int ldD,
                                           __bf16* G, int ldG, int col0, int lane) {
  int rw = (lane >> 4) * 8, cl = col0 + (lane & 15);
#pragma unroll
  for (int r = 0; r < 8; ++r) {
    float g = c[r] * (float)Dv[(rw + r) * ldD + cl];
    G[(rw + r) * ldG + cl] = (__bf16)g;
  }
}

// ---------------------------------------------------------------- AEV fwd
__global__ void radial_fwd(const float* __restrict__ x, const int* __restrict__ sp,
                           const int* __restrict__ pairs, int P,
                           float* __restrict__ aev) {
  int p = blockIdx.x * blockDim.x + threadIdx.x;
  if (p >= P) return;
  int i = pairs[p], j = pairs[P + p];
  float dx = x[3 * j] - x[3 * i], dy = x[3 * j + 1] - x[3 * i + 1],
        dz = x[3 * j + 2] - x[3 * i + 2];
  float d = sqrtf(dx * dx + dy * dy + dz * dz);
  float fc = cutoff(d, 5.2f);
  float* ai = aev + (size_t)i * D0P + sp[j] * 16;
  float* aj = aev + (size_t)j * D0P + sp[i] * 16;
#pragma unroll
  for (int k = 0; k < 16; ++k) {
    float t = d - (0.8f + 0.275f * k);
    float r = 0.25f * __expf(-19.7f * t * t) * fc;
    atomicAdd(ai + k, r);
    atomicAdd(aj + k, r);
  }
}

__global__ void angular_fwd(const float* __restrict__ x, const int* __restrict__ sp,
                            const int* __restrict__ trip, int T,
                            float* __restrict__ aev) {
  int t = blockIdx.x * blockDim.x + threadIdx.x;
  if (t >= T) return;
  int c = trip[t], a = trip[T + t], b = trip[2 * T + t];
  float v1x = x[3 * a] - x[3 * c], v1y = x[3 * a + 1] - x[3 * c + 1],
        v1z = x[3 * a + 2] - x[3 * c + 2];
  float v2x = x[3 * b] - x[3 * c], v2y = x[3 * b + 1] - x[3 * c + 1],
        v2z = x[3 * b + 2] - x[3 * c + 2];
  float d1 = sqrtf(v1x * v1x + v1y * v1y + v1z * v1z);
  float d2 = sqrtf(v2x * v2x + v2y * v2y + v2z * v2z);
  float dot = v1x * v2x + v1y * v2y + v1z * v2z;
  float cosang = 0.95f * dot / (d1 * d2);
  float ang = acosf(cosang);
  float fcj = cutoff(d1, 3.5f) * cutoff(d2, 3.5f);
  int sa = sp[a], sb = sp[b];
  int lo = sa < sb ? sa : sb, hi = sa < sb ? sb : sa;
  int pidx = lo * NS - (lo * (lo - 1)) / 2 + (hi - lo);
  float* dst = aev + (size_t)c * D0P + 112 + pidx * 32;
  float f1[4];
#pragma unroll
  for (int z = 0; z < 4; ++z) {
    float base = (1.0f + __cosf(ang - (z + 0.5f) * (PI_F / 4.0f))) * 0.5f;
    f1[z] = powf(base, 14.1f);
  }
#pragma unroll
  for (int q = 0; q < 8; ++q) {
    float u = (d1 + d2) * 0.5f - (0.8f + 0.3375f * q);
    float f2 = __expf(-12.5f * u * u);
#pragma unroll
    for (int z = 0; z < 4; ++z) atomicAdd(dst + q * 4 + z, 2.0f * f2 * f1[z] * fcj);
  }
}

__global__ void cvt_aev(const float* __restrict__ a, __bf16* __restrict__ o, int n) {
  int i = blockIdx.x * blockDim.x + threadIdx.x;
  if (i < n) o[i] = (__bf16)a[i];
}

// ---------------------------------------------------------------- species grouping
// counts -> tile-aligned bases + tile->species table (parallel, deterministic)
__global__ void count_kernel(const int* __restrict__ sgp, int N,
                             int* __restrict__ meta /* [0]=ntiles,[1..7]=bases */,
                             int* __restrict__ tileSpec) {
  __shared__ int cnt[NS];
  int tid = threadIdx.x;
  if (tid < NS) cnt[tid] = 0;
  __syncthreads();
  for (int n = tid; n < N; n += blockDim.x) {
    int s = sgp[n];
    if (s >= 0) atomicAdd(&cnt[s], 1);
  }
  __syncthreads();
  if (tid == 0) {
    int pos = 0;
    for (int s = 0; s < NS; ++s) {
      meta[1 + s] = pos;
      int padded = (cnt[s] + 15) & ~15;
      for (int tt = 0; tt < (padded >> 4); ++tt) tileSpec[(pos >> 4) + tt] = s;
      pos += padded;
    }
    meta[0] = pos >> 4;
  }
}

// stable rank within species (prefix count) -> deterministic scatter
__global__ void rank_scatter(const int* __restrict__ sgp, int N,
                             const int* __restrict__ meta, int* __restrict__ sorted) {
  int n = blockIdx.x * blockDim.x + threadIdx.x;
  if (n >= N) return;
  int s = sgp[n];
  if (s < 0) return;
  int r = 0;
  for (int m = 0; m < n; ++m) r += (sgp[m] == s) ? 1 : 0;
  sorted[meta[1 + s] + r] = n;
}

// ---------------------------------------------------------------- weight pack
// Output layout: [ms][ct][kc][lane][16 bf16]  (B fragment order for 16x16x32 bf16 WMMA)
__global__ void pack_kernel(const float* __restrict__ W, __bf16* __restrict__ out,
                            int R0, int C0, int RP, int CP, int trans, int total) {
  int i = blockIdx.x * blockDim.x + threadIdx.x;
  if (i >= total) return;
  int e = i & 15;
  int t = i >> 4;
  int lane = t & 31; t >>= 5;
  int kcn = RP / 32;
  int kc = t % kcn; t /= kcn;
  int ctn = CP / 16;
  int ct = t % ctn;
  int ms = t / ctn;
  int k = kc * 32 + (lane >> 4) * 8 + (e & 7) + ((e >> 3) << 4);
  int c = ct * 16 + (lane & 15);
  int Rl = trans ? C0 : R0, Cl = trans ? R0 : C0;
  float v = 0.0f;
  if (k < Rl && c < Cl)
    v = trans ? W[(size_t)ms * R0 * C0 + (size_t)c * C0 + k]
              : W[(size_t)ms * R0 * C0 + (size_t)k * C0 + c];
  out[i] = (__bf16)v;
}

// ---------------------------------------------------------------- fused MLP fwd+bwd
__global__ __launch_bounds__(256) void mlp_fused(
    const __bf16* __restrict__ aevb, const int* __restrict__ sorted,
    const int* __restrict__ tileSpec, const int* __restrict__ meta,
    const __bf16* __restrict__ W1p, const __bf16* __restrict__ W2p,
    const __bf16* __restrict__ W3p, const __bf16* __restrict__ W1Tp,
    const __bf16* __restrict__ W2Tp, const __bf16* __restrict__ W3Tp,
    const float* __restrict__ b1, const float* __restrict__ b2,
    const float* __restrict__ b3, const float* __restrict__ W4,
    const float* __restrict__ b4, float* __restrict__ gaev,
    float* __restrict__ Eout) {
  int tile = blockIdx.x, m = blockIdx.y;
  if (tile >= meta[0]) return;
  int s = tileSpec[tile];
  int ms = m * NS + s;
  int tid = threadIdx.x, lane = tid & 31, w = tid >> 5;

  extern __shared__ __align__(128) char smem[];
  __bf16* sAev = (__bf16*)(smem + SM_AEV);
  __bf16* sH1  = (__bf16*)(smem + SM_H1);
  __bf16* sD1  = (__bf16*)(smem + SM_D1);
  __bf16* sH2  = (__bf16*)(smem + SM_H2);
  __bf16* sD2  = (__bf16*)(smem + SM_D2);
  __bf16* sH3  = (__bf16*)(smem + SM_H3);
  __bf16* sD3  = (__bf16*)(smem + SM_D3);
  __bf16* sGA  = (__bf16*)(smem + SM_GA);
  __bf16* sGB  = (__bf16*)(smem + SM_GB);
  int*    sRows = (int*)(smem + SM_ROWS);
  float*  sEp   = (float*)(smem + SM_EP);

  if (tid < 16) sRows[tid] = sorted[tile * 16 + tid];
  __syncthreads();

  {  // stage A tile (16 rows x 2048B) into LDS via CDNA5 async copies
    unsigned ldsBase = (unsigned)(unsigned long long)(uintptr_t)(void*)sAev;
#pragma unroll
    for (int it = 0; it < 8; ++it) {
      int idx = it * 256 + tid;  // 2048 chunks of 16B
      int row = idx >> 7, ch = idx & 127;
      int atom = sRows[row];
      const __bf16* gp = aevb + (size_t)(atom < 0 ? 0 : atom) * D0P + ch * 8;
      unsigned lds = ldsBase + (unsigned)idx * 16u;
      unsigned long long ga = (unsigned long long)(uintptr_t)(const void*)gp;
      asm volatile("global_load_async_to_lds_b128 %0, %1, off"
                   :: "v"(lds), "v"(ga) : "memory");
    }
    asm volatile("s_wait_asynccnt 0x0" ::: "memory");
  }
  __syncthreads();

  const __bf16* aRow = sAev + (size_t)(lane & 15) * D0P;

  {  // L1: 1024 -> 256
    const __bf16* slab = W1p + (size_t)ms * (16 * 32 * 512);
    const float* bb = b1 + (size_t)ms * H1S;
    for (int ct = w; ct < 16; ct += 8) {
      v8f c = {};
      c = gemm16<32>(aRow, slab + (size_t)ct * (32 * 512), lane, c);
      store_act(c, bb[ct * 16 + (lane & 15)], sH1, sD1, H1S, ct * 16, lane);
    }
  }
  __syncthreads();
  {  // L2: 256 -> 192
    const __bf16* slab = W2p + (size_t)ms * (12 * 8 * 512);
    const float* bb = b2 + (size_t)ms * H2S;
    const __bf16* r1 = sH1 + (lane & 15) * H1S;
    for (int ct = w; ct < 12; ct += 8) {
      v8f c = {};
      c = gemm16<8>(r1, slab + (size_t)ct * (8 * 512), lane, c);
      store_act(c, bb[ct * 16 + (lane & 15)], sH2, sD2, H2S, ct * 16, lane);
    }
  }
  __syncthreads();
  {  // L3: 192 -> 160
    const __bf16* slab = W3p + (size_t)ms * (10 * 6 * 512);
    const float* bb = b3 + (size_t)ms * H3S;
    const __bf16* r2 = sH2 + (lane & 15) * H2S;
    for (int ct = w; ct < 10; ct += 8) {
      v8f c = {};
      c = gemm16<6>(r2, slab + (size_t)ct * (6 * 512), lane, c);
      store_act(c, bb[ct * 16 + (lane & 15)], sH3, sD3, H3S, ct * 16, lane);
    }
  }
  __syncthreads();
  {  // L4: atomic energies (ensemble mean, one atomic per block) + g3 = W4 * celu'(z3)
    const float* w4 = W4 + (size_t)ms * H3S;
    for (int idx = tid; idx < 16 * H3S; idx += 256) {
      int r = idx / H3S, k = idx - r * H3S;
      sGA[r * H3S + k] = (__bf16)(w4[k] * (float)sD3[r * H3S + k]);
    }
    if (tid < 16) {
      int atom = sRows[tid];
      float e = 0.0f;
      if (atom >= 0) {
        e = b4[ms];
        for (int k = 0; k < H3S; ++k) e += (float)sH3[tid * H3S + k] * w4[k];
      }
      sEp[tid] = e;
    }
    __syncthreads();
    if (tid == 0) {
      float e = 0.0f;
      for (int r = 0; r < 16; ++r) e += sEp[r];
      atomicAdd(Eout, e * (1.0f / ME));
    }
  }
  __syncthreads();
  {  // g2 = (g3 @ W3^T) * d2
    const __bf16* slab = W3Tp + (size_t)ms * (12 * 5 * 512);
    const __bf16* rg = sGA + (lane & 15) * H3S;
    for (int ct = w; ct < 12; ct += 8) {
      v8f c = {};
      c = gemm16<5>(rg, slab + (size_t)ct * (5 * 512), lane, c);
      store_grad(c, sD2, H2S, sGB, H2S, ct * 16, lane);
    }
  }
  __syncthreads();
  {  // g1 = (g2 @ W2^T) * d1
    const __bf16* slab = W2Tp + (size_t)ms * (16 * 6 * 512);
    const __bf16* rg = sGB + (lane & 15) * H2S;
    for (int ct = w; ct < 16; ct += 8) {
      v8f c = {};
      c = gemm16<6>(rg, slab + (size_t)ct * (6 * 512), lane, c);
      store_grad(c, sD1, H1S, sGA, H1S, ct * 16, lane);
    }
  }
  __syncthreads();
  {  // dE/daev += (g1 @ W1^T) / ME
    const __bf16* slab = W1Tp + (size_t)ms * (64 * 8 * 512);
    const __bf16* rg = sGA + (lane & 15) * H1S;
    int rw = (lane >> 4) * 8, cl = lane & 15;
    for (int ct = w; ct < 64; ct += 8) {
      v8f c = {};
      c = gemm16<8>(rg, slab + (size_t)ct * (8 * 512), lane, c);
      int col = ct * 16 + cl;
#pragma unroll
      for (int r = 0; r < 8; ++r) {
        int atom = sRows[rw + r];
        if (atom >= 0)
          atomicAdd(&gaev[(size_t)atom * D0P + col], c[r] * (1.0f / ME));
      }
    }
  }
}

// ---------------------------------------------------------------- SAE (block-reduced)
__global__ void sae_kernel(const int* __restrict__ sgp, const float* __restrict__ sae,
                           int N, float* __restrict__ Eout) {
  __shared__ float red[256];
  int tid = threadIdx.x;
  int n = blockIdx.x * blockDim.x + tid;
  float v = 0.0f;
  if (n < N) {
    int s = sgp[n];
    if (s >= 0) v = sae[s];
  }
  red[tid] = v;
  __syncthreads();
  for (int o = 128; o > 0; o >>= 1) {
    if (tid < o) red[tid] += red[tid + o];
    __syncthreads();
  }
  if (tid == 0) atomicAdd(Eout, red[0]);
}

// ---------------------------------------------------------------- AEV bwd
__global__ void radial_bwd(const float* __restrict__ x, const int* __restrict__ sp,
                           const int* __restrict__ pairs, int P,
                           const float* __restrict__ gaev, float* __restrict__ force) {
  int p = blockIdx.x * blockDim.x + threadIdx.x;
  if (p >= P) return;
  int i = pairs[p], j = pairs[P + p];
  float dx = x[3 * j] - x[3 * i], dy = x[3 * j + 1] - x[3 * i + 1],
        dz = x[3 * j + 2] - x[3 * i + 2];
  float d = sqrtf(dx * dx + dy * dy + dz * dz);
  float fc = cutoff(d, 5.2f), dfc = dcutoff(d, 5.2f);
  const float* gi = gaev + (size_t)i * D0P + sp[j] * 16;
  const float* gj = gaev + (size_t)j * D0P + sp[i] * 16;
  float gd = 0.0f;
#pragma unroll
  for (int k = 0; k < 16; ++k) {
    float t = d - (0.8f + 0.275f * k);
    float ex = 0.25f * __expf(-19.7f * t * t);
    float drdd = ex * (fc * (-39.4f * t) + dfc);
    gd += (gi[k] + gj[k]) * drdd;
  }
  float inv = 1.0f / d;
  float fx = gd * dx * inv, fy = gd * dy * inv, fz = gd * dz * inv;
  atomicAdd(&force[3 * j + 0], -fx);
  atomicAdd(&force[3 * j + 1], -fy);
  atomicAdd(&force[3 * j + 2], -fz);
  atomicAdd(&force[3 * i + 0], fx);
  atomicAdd(&force[3 * i + 1], fy);
  atomicAdd(&force[3 * i + 2], fz);
}

__global__ void angular_bwd(const float* __restrict__ x, const int* __restrict__ sp,
                            const int* __restrict__ trip, int T,
                            const float* __restrict__ gaev, float* __restrict__ force) {
  int t = blockIdx.x * blockDim.x + threadIdx.x;
  if (t >= T) return;
  int c = trip[t], a = trip[T + t], b = trip[2 * T + t];
  float v1x = x[3 * a] - x[3 * c], v1y = x[3 * a + 1] - x[3 * c + 1],
        v1z = x[3 * a + 2] - x[3 * c + 2];
  float v2x = x[3 * b] - x[3 * c], v2y = x[3 * b + 1] - x[3 * c + 1],
        v2z = x[3 * b + 2] - x[3 * c + 2];
  float d1 = sqrtf(v1x * v1x + v1y * v1y + v1z * v1z);
  float d2 = sqrtf(v2x * v2x + v2y * v2y + v2z * v2z);
  float dot = v1x * v2x + v1y * v2y + v1z * v2z;
  float inv12 = 1.0f / (d1 * d2);
  float cosang = 0.95f * dot * inv12;
  float ang = acosf(cosang);
  float fc1 = cutoff(d1, 3.5f), fc2 = cutoff(d2, 3.5f);
  float fcj = fc1 * fc2;
  float dfc1 = dcutoff(d1, 3.5f) * fc2, dfc2 = fc1 * dcutoff(d2, 3.5f);
  int sa = sp[a], sb = sp[b];
  int lo = sa < sb ? sa : sb, hi = sa < sb ? sb : sa;
  int pidx = lo * NS - (lo * (lo - 1)) / 2 + (hi - lo);
  const float* g = gaev + (size_t)c * D0P + 112 + pidx * 32;
  float f1[4], df1[4];
#pragma unroll
  for (int z = 0; z < 4; ++z) {
    float dz = ang - (z + 0.5f) * (PI_F / 4.0f);
    float base = (1.0f + __cosf(dz)) * 0.5f;
    float p13 = powf(base, 13.1f);
    f1[z] = p13 * base;
    df1[z] = 14.1f * p13 * (-0.5f * __sinf(dz));
  }
  float gd1 = 0.0f, gd2 = 0.0f, gang = 0.0f;
#pragma unroll
  for (int q = 0; q < 8; ++q) {
    float u = (d1 + d2) * 0.5f - (0.8f + 0.3375f * q);
    float f2 = __expf(-12.5f * u * u);
    float df2 = f2 * (-12.5f * u);
#pragma unroll
    for (int z = 0; z < 4; ++z) {
      float gv = 2.0f * g[q * 4 + z];
      gd1 += gv * (df2 * f1[z] * fcj + f2 * f1[z] * dfc1);
      gd2 += gv * (df2 * f1[z] * fcj + f2 * f1[z] * dfc2);
      gang += gv * f2 * df1[z] * fcj;
    }
  }
  float s2 = 1.0f - cosang * cosang;
  float gcos = gang * (-rsqrtf(fmaxf(s2, 1e-12f)));
  float k1 = gcos * 0.95f * inv12;
  float k2 = gcos * 0.95f * dot * inv12;
  float gv1x = gd1 * v1x / d1 + k1 * v2x - k2 * v1x / (d1 * d1);
  float gv1y = gd1 * v1y / d1 + k1 * v2y - k2 * v1y / (d1 * d1);
  float gv1z = gd1 * v1z / d1 + k1 * v2z - k2 * v1z / (d1 * d1);
  float gv2x = gd2 * v2x / d2 + k1 * v1x - k2 * v2x / (d2 * d2);
  float gv2y = gd2 * v2y / d2 + k1 * v1y - k2 * v2y / (d2 * d2);
  float gv2z = gd2 * v2z / d2 + k1 * v1z - k2 * v2z / (d2 * d2);
  atomicAdd(&force[3 * a + 0], -gv1x);
  atomicAdd(&force[3 * a + 1], -gv1y);
  atomicAdd(&force[3 * a + 2], -gv1z);
  atomicAdd(&force[3 * b + 0], -gv2x);
  atomicAdd(&force[3 * b + 1], -gv2y);
  atomicAdd(&force[3 * b + 2], -gv2z);
  atomicAdd(&force[3 * c + 0], gv1x + gv2x);
  atomicAdd(&force[3 * c + 1], gv1y + gv2y);
  atomicAdd(&force[3 * c + 2], gv1z + gv2z);
}

// ---------------------------------------------------------------- launch
extern "C" void kernel_launch(void* const* d_in, const int* in_sizes, int n_in,
                              void* d_out, int out_size, void* d_ws, size_t ws_size,
                              hipStream_t stream) {
  (void)n_in; (void)ws_size;
  const int* species = (const int*)d_in[0];
  const float* coords = (const float*)d_in[1];
  const int* pairs = (const int*)d_in[2];
  const int* trip = (const int*)d_in[3];
  const int* sgp = (const int*)d_in[4];
  const float* W1 = (const float*)d_in[5];
  const float* b1 = (const float*)d_in[6];
  const float* W2 = (const float*)d_in[7];
  const float* b2 = (const float*)d_in[8];
  const float* W3 = (const float*)d_in[9];
  const float* b3 = (const float*)d_in[10];
  const float* W4 = (const float*)d_in[11];
  const float* b4 = (const float*)d_in[12];
  const float* sae = (const float*)d_in[13];

  int N = in_sizes[0];
  int P = in_sizes[2] / 2;
  int T = in_sizes[3] / 3;
  int maxt = N / 16 + NS + 1;
  const int MS = ME * NS;

  char* w = (char*)d_ws;
  size_t off = 0;
  auto alloc = [&](size_t bytes) -> char* {
    char* p = w + off;
    off = (off + bytes + 255) & ~(size_t)255;
    return p;
  };
  float* aev = (float*)alloc((size_t)N * D0P * 4);
  __bf16* aevb = (__bf16*)alloc((size_t)N * D0P * 2);
  float* gaev = (float*)alloc((size_t)N * D0P * 4);
  int* sorted = (int*)alloc((size_t)maxt * 16 * 4);
  int* tileSpec = (int*)alloc((size_t)maxt * 4);
  int* meta = (int*)alloc(64);
  __bf16* W1p  = (__bf16*)alloc((size_t)MS * 16 * 32 * 512 * 2);
  __bf16* W1Tp = (__bf16*)alloc((size_t)MS * 64 * 8 * 512 * 2);
  __bf16* W2p  = (__bf16*)alloc((size_t)MS * 12 * 8 * 512 * 2);
  __bf16* W2Tp = (__bf16*)alloc((size_t)MS * 16 * 6 * 512 * 2);
  __bf16* W3p  = (__bf16*)alloc((size_t)MS * 10 * 6 * 512 * 2);
  __bf16* W3Tp = (__bf16*)alloc((size_t)MS * 12 * 5 * 512 * 2);

  hipMemsetAsync(aev, 0, (size_t)N * D0P * 4, stream);
  hipMemsetAsync(gaev, 0, (size_t)N * D0P * 4, stream);
  hipMemsetAsync(d_out, 0, (size_t)out_size * 4, stream);
  hipMemsetAsync(sorted, 0xFF, (size_t)maxt * 16 * 4, stream);
  hipMemsetAsync(tileSpec, 0, (size_t)maxt * 4, stream);

  count_kernel<<<1, 256, 0, stream>>>(sgp, N, meta, tileSpec);
  rank_scatter<<<(N + 255) / 256, 256, 0, stream>>>(sgp, N, meta, sorted);
  radial_fwd<<<(P + 255) / 256, 256, 0, stream>>>(coords, species, pairs, P, aev);
  angular_fwd<<<(T + 255) / 256, 256, 0, stream>>>(coords, species, trip, T, aev);
  int nel = N * D0P;
  cvt_aev<<<(nel + 255) / 256, 256, 0, stream>>>(aev, aevb, nel);

  int tot;
  tot = MS * 16 * 32 * 512;
  pack_kernel<<<(tot + 255) / 256, 256, 0, stream>>>(W1, W1p, 1008, 256, 1024, 256, 0, tot);
  tot = MS * 64 * 8 * 512;
  pack_kernel<<<(tot + 255) / 256, 256, 0, stream>>>(W1, W1Tp, 1008, 256, 256, 1024, 1, tot);
  tot = MS * 12 * 8 * 512;
  pack_kernel<<<(tot + 255) / 256, 256, 0, stream>>>(W2, W2p, 256, 192, 256, 192, 0, tot);
  tot = MS * 16 * 6 * 512;
  pack_kernel<<<(tot + 255) / 256, 256, 0, stream>>>(W2, W2Tp, 256, 192, 192, 256, 1, tot);
  tot = MS * 10 * 6 * 512;
  pack_kernel<<<(tot + 255) / 256, 256, 0, stream>>>(W3, W3p, 192, 160, 192, 160, 0, tot);
  tot = MS * 12 * 5 * 512;
  pack_kernel<<<(tot + 255) / 256, 256, 0, stream>>>(W3, W3Tp, 192, 160, 160, 192, 1, tot);

  hipFuncSetAttribute((const void*)mlp_fused,
                      hipFuncAttributeMaxDynamicSharedMemorySize, SM_TOTAL);
  dim3 grid(maxt, ME);
  mlp_fused<<<grid, 256, SM_TOTAL, stream>>>(aevb, sorted, tileSpec, meta, W1p, W2p,
                                             W3p, W1Tp, W2Tp, W3Tp, b1, b2, b3, W4,
                                             b4, gaev, (float*)d_out);
  sae_kernel<<<(N + 255) / 256, 256, 0, stream>>>(sgp, sae, N, (float*)d_out);

  float* force = (float*)d_out + 1;
  radial_bwd<<<(P + 255) / 256, 256, 0, stream>>>(coords, species, pairs, P, gaev, force);
  angular_bwd<<<(T + 255) / 256, 256, 0, stream>>>(coords, species, trip, T, gaev, force);
}
